// AttMultiHeadMultiResLoc_87832081203822
// MI455X (gfx1250) — matrix operand order repaired and checked by reading, not compiled, not run on hardware
//
#include <hip/hip_runtime.h>
#include <math.h>

#define BB 32
#define TT 1500
#define EE 512
#define DD 1024
#define KK 512
#define VV 512
#define HH 4
#define CCH 64
#define HK (HH*KK)   /* 2048 combined head*k columns */
#define SCALING 0.04419417382415922f  /* 1/sqrt(512) */

typedef __bf16 v16bf __attribute__((ext_vector_type(16)));
typedef float v8f __attribute__((ext_vector_type(8)));

union FragU { uint4 u[2]; v16bf v; };

__device__ __forceinline__ unsigned short f2bf(float x) {
  unsigned u = __float_as_uint(x);
  unsigned r = u + 0x7fffu + ((u >> 16) & 1u);
  return (unsigned short)(r >> 16);
}

/* async 16B global -> LDS copy (CDNA5 ASYNCcnt path, no VGPR round-trip) */
__device__ __forceinline__ void async_copy16(unsigned lds_off, const void* gptr) {
  asm volatile("global_load_async_to_lds_b128 %0, %1, off"
               :: "v"(lds_off), "v"((unsigned long long)(uintptr_t)gptr)
               : "memory");
}

__device__ __forceinline__ void wait_async0() {
#if __has_builtin(__builtin_amdgcn_s_wait_asynccnt)
  __builtin_amdgcn_s_wait_asynccnt(0);
#else
  asm volatile("s_wait_asynccnt 0x0" ::: "memory");
#endif
}

/* ---- fp32 -> bf16 conversion of enc_hs_pad (L2-resident afterwards) ---- */
__global__ void k_cvt_enc(const float* __restrict__ src,
                          unsigned short* __restrict__ dst, int n) {
  for (int i = blockIdx.x * blockDim.x + threadIdx.x; i < n;
       i += gridDim.x * blockDim.x)
    dst[i] = f2bf(src[i]);
}

/* ---- Wk[h][e][k] -> WkT_bf16[h*K + k][e] (transposed for B-fragments) ---- */
__global__ void k_cvt_wkT(const float* __restrict__ Wk,
                          unsigned short* __restrict__ dst) {
  int i = blockIdx.x * blockDim.x + threadIdx.x;   /* over 2048*512 */
  int e = i & 511;
  int hk = i >> 9;
  int h = hk >> 9;
  int k = hk & 511;
  dst[i] = f2bf(Wk[((size_t)(h * EE + e)) * KK + k]);
}

/* ---- q[b][h*K+k] = dec_z[b] . Wq[h][:,k] + bq ---- */
__global__ void k_q(const float* __restrict__ dz, const float* __restrict__ Wq,
                    const float* __restrict__ bq, float* __restrict__ q) {
  int g = blockIdx.x * blockDim.x + threadIdx.x;   /* B*2048 */
  int b = g >> 11;
  int hk = g & 2047;
  int h = hk >> 9;
  int k = hk & 511;
  float acc = bq[hk];
  const float* z = dz + (size_t)b * DD;
  const float* W = Wq + (size_t)h * DD * KK + k;
  for (int e = 0; e < DD; e++) acc += z[e] * W[(size_t)e * KK];
  q[g] = acc;
}

/* ---- exclusive prefix sums of conv filter taps per (h,c) ---- */
__global__ void k_wpre(const float* __restrict__ w0, const float* __restrict__ w1,
                       const float* __restrict__ w2, const float* __restrict__ w3,
                       float* __restrict__ Wpre) {
  int tid = threadIdx.x;             /* 256 = H*CH */
  int h = tid >> 6, c = tid & 63;
  const float* w = (h == 0) ? w0 : (h == 1) ? w1 : (h == 2) ? w2 : w3;
  int af = (100 * (h + 1)) / 4;
  int ksz = 2 * af + 1;
  float* P = Wpre + (size_t)tid * 256;
  float s = 0.f;
  P[0] = 0.f;
  for (int j = 0; j < ksz; j++) { s += w[c * ksz + j]; P[j + 1] = s; }
}

/* ---- G[h][j][k] = sum_c Wpre[h][c][j] * Watt[h][c][k] ---- */
__global__ void k_g(const float* __restrict__ Wpre, const float* __restrict__ Watt,
                    float* __restrict__ G) {
  int h = blockIdx.x;
  int j = blockIdx.y;
  int k = threadIdx.x;
  int af = (100 * (h + 1)) / 4;
  int ksz = 2 * af + 1;
  if (j > ksz) return;
  float acc = 0.f;
  for (int c = 0; c < CCH; c++)
    acc += Wpre[((size_t)(h * 64 + c)) * 256 + j] *
           Watt[((size_t)(h * 64 + c)) * KK + k];
  G[((size_t)(h * 256 + j)) * KK + k] = acc;
}

/* ---- fused WMMA kernel: logits z[h][b][t] = SCALING*(gw.tanh(enc.Wk + ack + q)+gb), masked ----
   B-tiles are double-buffered in LDS and staged with global_load_async_to_lds_b128. */
__global__ __launch_bounds__(256) void k_e(
    const unsigned short* __restrict__ encb, const unsigned short* __restrict__ wkT,
    const float* __restrict__ q, const float* __restrict__ G,
    const float* __restrict__ gw, const float* __restrict__ gb,
    const int* __restrict__ lens, float* __restrict__ z) {
  __shared__ unsigned short ldsB[2][16 * 520];   /* 16 k-rows x 512 e, +8 pad vs bank conflicts */
  int b = blockIdx.x;
  int tid = threadIdx.x;
  int wave = tid >> 5, lane = tid & 31;
  int half = lane >> 4, lq = lane & 15;
  int t0 = blockIdx.y * 128 + wave * 16;      /* each wave owns one 16-row t-tile */
  int len = lens[b];
  float invlen = 1.0f / (float)len;

  /* Preload A fragments (16 t-rows x all 512 e) into registers; ISA 16-bit A layout:
     lane: M = lq, VGPR0-3 = K (half*8 + 0..7), VGPR4-7 = K (16 + half*8 + 0..7). */
  int trow = t0 + lq; if (trow > TT - 1) trow = TT - 1;
  const uint4* abase = (const uint4*)(encb + ((size_t)b * TT + trow) * EE);
  uint4 afrag[32];
#pragma unroll
  for (int s = 0; s < 16; s++) {
    afrag[2 * s]     = abase[s * 4 + half];
    afrag[2 * s + 1] = abase[s * 4 + half + 2];
  }

  /* async stage of B tile n into LDS buffer bufid: WkT rows [n*16, n*16+16) x 512 e */
  auto stage = [&](int n, int bufid) {
    const unsigned short* wsrc = wkT + (size_t)n * 16 * EE;
#pragma unroll
    for (int rep = 0; rep < 4; rep++) {
      int c = tid + rep * 256;
      int row = c >> 6, col8 = (c & 63) * 8;
      unsigned loff = (unsigned)(uintptr_t)&ldsB[bufid][row * 520 + col8];
      async_copy16(loff, wsrc + row * EE + col8);
    }
  };

  stage(0, 0);

  float eacc[8];
  for (int n = 0; n < 128; n++) {             /* 128 N-tiles = 4 heads x 32 k-tiles */
    int head = n >> 5, kt = n & 31;
    int cur = n & 1;
    if (kt == 0) {
#pragma unroll
      for (int i = 0; i < 8; i++) eacc[i] = 0.f;
    }
    wait_async0();          /* our async writes for tile n landed            */
    __syncthreads();        /* everyone's landed; prev readers of buf cur^1 done */
    if (n + 1 < 128) stage(n + 1, cur ^ 1);

    v8f acc = {0.f, 0.f, 0.f, 0.f, 0.f, 0.f, 0.f, 0.f};
#pragma unroll
    for (int s = 0; s < 16; s++) {            /* reduce E=512 in steps of 32 */
      FragU a, bb;
      a.u[0] = afrag[2 * s];
      a.u[1] = afrag[2 * s + 1];
      /* B layout: lane: N = lq, K = half*8 + 0..7, then 16 + half*8 + 0..7 */
      int e0 = s * 32 + half * 8;
      bb.u[0] = *(const uint4*)&ldsB[cur][lq * 520 + e0];
      bb.u[1] = *(const uint4*)&ldsB[cur][lq * 520 + e0 + 16];
      acc = __builtin_amdgcn_wmma_f32_16x16x32_bf16(false, a.v, false, bb.v,
                                                    (short)0, acc, false, false);
    }

    /* epilogue: acc lane layout: N = lq, rows M = half*8 + i */
    int kk = head * KK + kt * 16 + lq;
    float qv = q[(size_t)b * HK + kk];
    float gwv = gw[kk];
    int af = (100 * (head + 1)) / 4;
    int ksz = 2 * af + 1;
    int p = af;
    const float* Gh = G + (size_t)head * 256 * KK;
    int kcol = kt * 16 + lq;
    bool interior = (t0 >= p) && (t0 + 15 <= len - 1 - p);
    if (interior) {
      /* full window: jlo=0 (G[0]=0), jhi=ksz */
      float av = Gh[(size_t)ksz * KK + kcol] * invlen;
#pragma unroll
      for (int i = 0; i < 8; i++)
        eacc[i] += tanhf(acc[i] + qv + av) * gwv;
    } else {
#pragma unroll
      for (int i = 0; i < 8; i++) {
        int t = t0 + half * 8 + i;
        int jlo = p - t; jlo = jlo < 0 ? 0 : jlo; jlo = jlo > ksz ? ksz : jlo;
        int jhi = len + p - t; jhi = jhi > ksz ? ksz : jhi; jhi = jhi < jlo ? jlo : jhi;
        float av = (Gh[(size_t)jhi * KK + kcol] - Gh[(size_t)jlo * KK + kcol]) * invlen;
        eacc[i] += tanhf(acc[i] + qv + av) * gwv;
      }
    }
    if (kt == 31) {                           /* head finished: reduce 16 columns, write */
      float gbh = gb[head];
#pragma unroll
      for (int i = 0; i < 8; i++) {
        float v = eacc[i];
        v += __shfl_xor(v, 1);
        v += __shfl_xor(v, 2);
        v += __shfl_xor(v, 4);
        v += __shfl_xor(v, 8);
        if (lq == 0) {
          int t = t0 + half * 8 + i;
          if (t < TT)
            z[((size_t)head * BB + b) * TT + t] =
                (t < len) ? SCALING * (v + gbh) : __int_as_float(0xff800000u);
        }
      }
    }
  }
}

/* ---- in-place softmax over T per (h,b) row ---- */
__global__ __launch_bounds__(256) void k_softmax(float* __restrict__ wsbuf) {
  __shared__ float buf[TT];
  __shared__ float red[256];
  int hb = blockIdx.x;
  float* row = wsbuf + (size_t)hb * TT;
  int tid = threadIdx.x;
  for (int i = tid; i < TT; i += 256) buf[i] = row[i];
  __syncthreads();
  float m = -3.4e38f;
  for (int i = tid; i < TT; i += 256) m = fmaxf(m, buf[i]);
  red[tid] = m; __syncthreads();
  for (int s = 128; s > 0; s >>= 1) {
    if (tid < s) red[tid] = fmaxf(red[tid], red[tid + s]);
    __syncthreads();
  }
  float M = red[0];
  __syncthreads();
  float ssum = 0.f;
  for (int i = tid; i < TT; i += 256) {
    float ex = __expf(buf[i] - M);
    buf[i] = ex;
    ssum += ex;
  }
  red[tid] = ssum; __syncthreads();
  for (int s = 128; s > 0; s >>= 1) {
    if (tid < s) red[tid] += red[tid + s];
    __syncthreads();
  }
  float inv = 1.f / red[0];
  for (int i = tid; i < TT; i += 256) row[i] = buf[i] * inv;
}

/* ---- u[h][b][e] = sum_t w[h][b][t] * enc[b][t][e] ---- */
__global__ __launch_bounds__(128) void k_u(const float* __restrict__ enc,
                                           const float* __restrict__ wbuf,
                                           float* __restrict__ u) {
  __shared__ float wl[4 * 256];
  int b = blockIdx.x;
  int e = blockIdx.y * 128 + threadIdx.x;
  float a0 = 0.f, a1 = 0.f, a2 = 0.f, a3 = 0.f;
  for (int t0 = 0; t0 < TT; t0 += 256) {
    __syncthreads();
#pragma unroll
    for (int r = 0; r < 8; r++) {
      int idx = threadIdx.x + r * 128;      /* 0..1023 */
      int h = idx >> 8, t = idx & 255;
      wl[idx] = (t0 + t < TT) ? wbuf[((size_t)h * BB + b) * TT + t0 + t] : 0.f;
    }
    __syncthreads();
    int tmax = TT - t0; if (tmax > 256) tmax = 256;
    for (int t = 0; t < tmax; t++) {
      float x = enc[((size_t)b * TT + t0 + t) * EE + e];
      a0 += x * wl[0 * 256 + t];
      a1 += x * wl[1 * 256 + t];
      a2 += x * wl[2 * 256 + t];
      a3 += x * wl[3 * 256 + t];
    }
  }
  u[((size_t)0 * BB + b) * EE + e] = a0;
  u[((size_t)1 * BB + b) * EE + e] = a1;
  u[((size_t)2 * BB + b) * EE + e] = a2;
  u[((size_t)3 * BB + b) * EE + e] = a3;
}

/* ---- c[h][b][v] = u[h][b][:] . Wv[h][:,v] ---- */
__global__ __launch_bounds__(512) void k_cv(const float* __restrict__ u,
                                            const float* __restrict__ Wv,
                                            float* __restrict__ cv) {
  __shared__ float ul[512];
  int hb = blockIdx.x;
  int h = hb >> 5;
  int v = threadIdx.x;
  ul[v] = u[(size_t)hb * EE + v];
  __syncthreads();
  float acc = 0.f;
  for (int e = 0; e < EE; e++) acc += ul[e] * Wv[((size_t)h * EE + e) * VV + v];
  cv[(size_t)hb * VV + v] = acc;
}

/* ---- out[b][ep] = concat_h(c[h][b]) . Wo ---- */
__global__ __launch_bounds__(512) void k_out(const float* __restrict__ cv,
                                             const float* __restrict__ Wo,
                                             float* __restrict__ outc) {
  __shared__ float cc[2048];
  int b = blockIdx.x;
  int ep = threadIdx.x;
#pragma unroll
  for (int r = 0; r < 4; r++) {
    int j = threadIdx.x + r * 512;
    cc[j] = cv[(((size_t)(j >> 9)) * BB + b) * VV + (j & 511)];
  }
  __syncthreads();
  float acc = 0.f;
  for (int j = 0; j < 2048; j++) acc += cc[j] * Wo[(size_t)j * EE + ep];
  outc[(size_t)b * EE + ep] = acc;
}

extern "C" void kernel_launch(void* const* d_in, const int* in_sizes, int n_in,
                              void* d_out, int out_size, void* d_ws, size_t ws_size,
                              hipStream_t stream) {
  (void)in_sizes; (void)n_in; (void)out_size; (void)ws_size;
  const float* enc  = (const float*)d_in[0];
  const int*   lens = (const int*)d_in[1];
  const float* dz   = (const float*)d_in[2];
  const float* Wq   = (const float*)d_in[3];
  const float* bq   = (const float*)d_in[4];
  const float* Wk   = (const float*)d_in[5];
  const float* Wv   = (const float*)d_in[6];
  const float* gw   = (const float*)d_in[7];
  const float* gb   = (const float*)d_in[8];
  const float* Watt = (const float*)d_in[9];
  const float* Wo   = (const float*)d_in[10];
  const float* cw0  = (const float*)d_in[11];
  const float* cw1  = (const float*)d_in[12];
  const float* cw2  = (const float*)d_in[13];
  const float* cw3  = (const float*)d_in[14];

  char* ws = (char*)d_ws;
  unsigned short* encb = (unsigned short*)(ws);              /* 49,152,000 B */
  unsigned short* wkT  = (unsigned short*)(ws + 49152000);   /*  2,097,152 B */
  float* qbuf = (float*)(ws + 51249152);                     /*    262,144 B */
  float* Wpre = (float*)(ws + 51511296);                     /*    262,144 B */
  float* G    = (float*)(ws + 51773440);                     /*  2,097,152 B */
  float* ubuf = (float*)(ws + 53870592);                     /*    262,144 B */
  float* cvb  = (float*)(ws + 54132736);                     /*    262,144 B */

  float* outc = (float*)d_out;                 /* [B, EPROJS]   */
  float* wbuf = (float*)d_out + BB * EE;       /* [H, B, T] ws  */

  hipLaunchKernelGGL(k_cvt_enc, dim3(4096), dim3(256), 0, stream, enc, encb, BB * TT * EE);
  hipLaunchKernelGGL(k_cvt_wkT, dim3(4096), dim3(256), 0, stream, Wk, wkT);
  hipLaunchKernelGGL(k_q, dim3(BB * HK / 256), dim3(256), 0, stream, dz, Wq, bq, qbuf);
  hipLaunchKernelGGL(k_wpre, dim3(1), dim3(256), 0, stream, cw0, cw1, cw2, cw3, Wpre);
  hipLaunchKernelGGL(k_g, dim3(HH, 202), dim3(512), 0, stream, Wpre, Watt, G);
  hipLaunchKernelGGL(k_e, dim3(BB, 12), dim3(256), 0, stream, encb, wkT, qbuf, G, gw, gb, lens, wbuf);
  hipLaunchKernelGGL(k_softmax, dim3(HH * BB), dim3(256), 0, stream, wbuf);
  hipLaunchKernelGGL(k_u, dim3(BB, 4), dim3(128), 0, stream, enc, wbuf, ubuf);
  hipLaunchKernelGGL(k_cv, dim3(HH * BB), dim3(512), 0, stream, ubuf, Wv, cvb);
  hipLaunchKernelGGL(k_out, dim3(BB), dim3(512), 0, stream, cvb, Wo, outc);
}
